// GAT_17300128268969
// MI455X (gfx1250) — matrix operand order repaired and checked by reading, not compiled, not run on hardware
//
#include <hip/hip_runtime.h>
#include <math.h>

// ---- problem constants (from the reference) ----
#define BB   8
#define NN   768
#define DD   512
#define HH   8
#define HID  256
#define HDm  64
#define MM   (BB * NN)          // 6144
#define G4   (4 * HID)          // 1024
#define NPAD 776                // 768 + 8 f32 pad -> stagger LDS banks

typedef __attribute__((ext_vector_type(16))) __bf16 v16bf;
typedef __attribute__((ext_vector_type(8)))  float  v8f;

// ---------------------------------------------------------------------------
// f32 -> bf16 conversion
// ---------------------------------------------------------------------------
__global__ void cvt_bf16_kernel(const float* __restrict__ in,
                                __bf16* __restrict__ out, int n) {
    int i = blockIdx.x * 256 + threadIdx.x;
    if (i < n) out[i] = (__bf16)in[i];
}

// ---------------------------------------------------------------------------
// WMMA GEMM with 32x64 register tile per wave (2 M-frags x 4 N-frags):
//   C[m,n] = sum_k A[m,k]*W[n,k] (+bias0[n]+bias1[n]) (+Cin[m,n])
// A: (Mrows x K) bf16 row-major (lda).  W: (Nout x K) bf16 row-major (ldw) —
// nn.Linear layout, so a B-fragment column is a contiguous row of W.
// Block = 128 threads (4 waves stacked along M): block tile = 128 x 64.
// Grid = (Nout/64, Mrows/128). All dims are exact multiples -> EXEC all-ones.
// Fragment reuse: 12 b128 loads per k-step feed 8 WMMAs (~21 FLOP/byte).
// ---------------------------------------------------------------------------
__global__ void gemm_wmma_kernel(const __bf16* __restrict__ A, int lda,
                                 const __bf16* __restrict__ W, int ldw,
                                 const float* __restrict__ bias0,
                                 const float* __restrict__ bias1,
                                 const float* __restrict__ Cin,
                                 float* __restrict__ Cf,
                                 __bf16* __restrict__ Cb,
                                 int ldc, int K) {
    const int lane = threadIdx.x & 31;
    const int wave = threadIdx.x >> 5;
    const int c0 = blockIdx.x * 64;
    const int r0 = blockIdx.y * 128 + wave * 32;

    // A fragment mapping (16-bit A 16x32): lanes 0-15 row m, K {ak..ak+15}
    const int arow = lane & 15;
    const int ak   = (lane >> 4) * 16;
    // B fragment mapping (16-bit B 32x16): lane holds col, 16 consecutive K
    const int bcol = lane & 15;
    const int bk   = (lane >> 4) * 16;

    v8f acc[2][4] = {};
    for (int k0 = 0; k0 < K; k0 += 32) {
        v16bf a[2], b[4];
#pragma unroll
        for (int i2 = 0; i2 < 2; ++i2) {
            const __bf16* pa = A + (size_t)(r0 + i2 * 16 + arow) * lda + k0 + ak;
#pragma unroll
            for (int i = 0; i < 8; ++i) { a[i2][i] = pa[i]; a[i2][8 + i] = pa[8 + i]; }
        }
#pragma unroll
        for (int t = 0; t < 4; ++t) {
            const __bf16* pb = W + (size_t)(c0 + t * 16 + bcol) * ldw + k0 + bk;
#pragma unroll
            for (int i = 0; i < 16; ++i) b[t][i] = pb[i];
        }
#pragma unroll
        for (int i2 = 0; i2 < 2; ++i2)
#pragma unroll
            for (int t = 0; t < 4; ++t)
                acc[i2][t] = __builtin_amdgcn_wmma_f32_16x16x32_bf16(
                    false, a[i2], false, b[t], (short)0, acc[i2][t], false, false);
    }

    const int cn = lane & 15;
    const int mo = (lane >> 4) * 8;
#pragma unroll
    for (int t = 0; t < 4; ++t) {
        const int col = c0 + t * 16 + cn;
        float bsum = 0.f;
        if (bias0) bsum += bias0[col];
        if (bias1) bsum += bias1[col];
#pragma unroll
        for (int i2 = 0; i2 < 2; ++i2) {
#pragma unroll
            for (int j = 0; j < 8; ++j) {
                int r = r0 + i2 * 16 + mo + j;
                float v = acc[i2][t][j] + bsum;
                size_t idx = (size_t)r * ldc + col;
                if (Cin) v += Cin[idx];
                if (Cf) Cf[idx] = v;
                if (Cb) Cb[idx] = (__bf16)v;
            }
        }
    }
}

// ---------------------------------------------------------------------------
// Fused masked multi-head attention for one (b, h, 16-query tile).
// head/tail/info: (B,N,D) bf16, head-sliced at h*64. Writes av (B,N,D) bf16.
// Block = 128 threads (4 waves). Score tile lives in LDS f32 (row-padded);
// softmax probabilities are re-materialized as bf16 A-fragments on the fly.
// ---------------------------------------------------------------------------
__global__ void attn_kernel(const __bf16* __restrict__ head,
                            const __bf16* __restrict__ tail,
                            const __bf16* __restrict__ info,
                            const int* __restrict__ graph,
                            __bf16* __restrict__ av) {
    __shared__ float ssc[16 * NPAD];   // masked/scaled scores, f32
    __shared__ float sred[16 * 8];
    __shared__ float srowmax[16];
    __shared__ float srowinv[16];

    const int blk = blockIdx.x;                 // ((b*H + h)*48 + qt)
    const int qt = blk % 48;
    const int h  = (blk / 48) % HH;
    const int b  = blk / (48 * HH);
    const int n0 = qt * 16;
    const int hofs = h * HDm;

    const int lane = threadIdx.x & 31;
    const int wave = threadIdx.x >> 5;
    const float inv_scale = 0.044194173824159216f;   // 1/sqrt(512)

    const int arow = lane & 15, ak = (lane >> 4) * 16;
    const int bcol = lane & 15, bk = (lane >> 4) * 16;
    const int cn = lane & 15,  mo = (lane >> 4) * 8;

    // ---- Q fragments for K=0..63 (hoisted; shared across all m-tiles) ----
    v16bf qa0, qa1;
    {
        const __bf16* p = head + (size_t)(b * NN + n0 + arow) * DD + hofs + ak;
#pragma unroll
        for (int i = 0; i < 8; ++i) {
            qa0[i] = p[i];      qa0[8 + i] = p[8 + i];
            qa1[i] = p[32 + i]; qa1[8 + i] = p[40 + i];
        }
    }

    // ---- scores: wave handles m-tiles wave, wave+4, ... (48 tiles total) ----
    for (int mt = wave; mt < 48; mt += 4) {
        int m0 = mt * 16;
        v16bf kb0, kb1;
        const __bf16* pb = tail + (size_t)(b * NN + m0 + bcol) * DD + hofs + bk;
#pragma unroll
        for (int i = 0; i < 16; ++i) { kb0[i] = pb[i]; kb1[i] = pb[32 + i]; }
        v8f sc = {};
        sc = __builtin_amdgcn_wmma_f32_16x16x32_bf16(false, qa0, false, kb0,
                                                     (short)0, sc, false, false);
        sc = __builtin_amdgcn_wmma_f32_16x16x32_bf16(false, qa1, false, kb1,
                                                     (short)0, sc, false, false);
#pragma unroll
        for (int j = 0; j < 8; ++j) {
            int r = mo + j;
            int m = m0 + cn;
            float s = sc[j] * inv_scale;
            if (graph[((size_t)b * NN + (n0 + r)) * NN + m] == 0) s = -INFINITY;
            ssc[r * NPAD + m] = s;
        }
    }
    __syncthreads();

    // ---- softmax stats: 8 threads per query row ----
    {
        const int row = threadIdx.x >> 3;
        const int sub = threadIdx.x & 7;
        float mx = -INFINITY;
        for (int c = sub; c < NN; c += 8) mx = fmaxf(mx, ssc[row * NPAD + c]);
        sred[row * 8 + sub] = mx;
        __syncthreads();
        if (sub == 0) {
            float m2 = sred[row * 8];
#pragma unroll
            for (int i = 1; i < 8; ++i) m2 = fmaxf(m2, sred[row * 8 + i]);
            srowmax[row] = m2;
        }
        __syncthreads();
        float rm = srowmax[row];
        float sm = 0.f;
        if (rm > -INFINITY)
            for (int c = sub; c < NN; c += 8) sm += __expf(ssc[row * NPAD + c] - rm);
        sred[row * 8 + sub] = sm;
        __syncthreads();
        if (sub == 0) {
            float s2 = 0.f;
#pragma unroll
            for (int i = 0; i < 8; ++i) s2 += sred[row * 8 + i];
            srowinv[row] = (s2 > 0.f) ? (1.f / s2) : 0.f;   // all-masked row -> zeros
        }
        __syncthreads();
    }

    // ---- AV: wave owns d-tile d0 = wave*16; K = m = 0..767 ----
    const int d0 = wave * 16;
    const float rmA = srowmax[arow];
    const float riA = srowinv[arow];
    v8f acc = {};
    for (int k0 = 0; k0 < NN; k0 += 32) {
        v16bf a, bv;
#pragma unroll
        for (int i = 0; i < 8; ++i) {
            float s0 = ssc[arow * NPAD + k0 + ak + i];
            float s1 = ssc[arow * NPAD + k0 + ak + 8 + i];
            a[i]     = (__bf16)((riA > 0.f) ? __expf(s0 - rmA) * riA : 0.f);
            a[8 + i] = (__bf16)((riA > 0.f) ? __expf(s1 - rmA) * riA : 0.f);
        }
        const __bf16* pb = info + (size_t)(b * NN + k0 + bk) * DD + hofs + d0 + bcol;
#pragma unroll
        for (int i = 0; i < 16; ++i) bv[i] = pb[(size_t)i * DD];
        acc = __builtin_amdgcn_wmma_f32_16x16x32_bf16(false, a, false, bv,
                                                      (short)0, acc, false, false);
    }
#pragma unroll
    for (int j = 0; j < 8; ++j)
        av[(size_t)(b * NN + n0 + mo + j) * DD + hofs + d0 + cn] = (__bf16)acc[j];
}

// ---------------------------------------------------------------------------
// LSTM cell elementwise: g (M x 1024, gates i|f|g|o), c0 (M x 256 or null).
// h = sigmoid(o) * tanh(sigmoid(f)*c + sigmoid(i)*tanh(g))
// ---------------------------------------------------------------------------
__global__ void lstm_elem_kernel(const float* __restrict__ g,
                                 const float* __restrict__ c0,
                                 float* __restrict__ hout) {
    int idx = blockIdx.x * 256 + threadIdx.x;
    if (idx >= MM * HID) return;
    int m = idx >> 8;
    int k = idx & 255;
    const float* gr = g + (size_t)m * G4;
    float gi = gr[k], gf = gr[HID + k], gg = gr[2 * HID + k], go = gr[3 * HID + k];
    float c  = c0 ? c0[idx] : 0.f;
    float si = 1.f / (1.f + __expf(-gi));
    float sf = 1.f / (1.f + __expf(-gf));
    float so = 1.f / (1.f + __expf(-go));
    float c2 = sf * c + si * tanhf(gg);
    hout[idx] = so * tanhf(c2);
}

// ---------------------------------------------------------------------------
// LayerNorm over rows of 512; one wave32 per row (16 elems/lane).
// Writes f32 (memory / d_out) and bf16 (GEMM operand) simultaneously.
// ---------------------------------------------------------------------------
__global__ void layernorm_kernel(const float* __restrict__ X,
                                 const float* __restrict__ gamma,
                                 const float* __restrict__ beta,
                                 float* __restrict__ Yf,
                                 __bf16* __restrict__ Yb) {
    int row  = blockIdx.x * 8 + (threadIdx.x >> 5);
    int lane = threadIdx.x & 31;
    const float* xr = X + (size_t)row * DD;
    float v[16];
    float s = 0.f, s2 = 0.f;
#pragma unroll
    for (int i = 0; i < 16; ++i) {
        float x = xr[lane + i * 32];
        v[i] = x; s += x; s2 += x * x;
    }
#pragma unroll
    for (int o = 16; o > 0; o >>= 1) {
        s  += __shfl_xor(s,  o, 32);
        s2 += __shfl_xor(s2, o, 32);
    }
    float mean = s * (1.f / DD);
    float var  = s2 * (1.f / DD) - mean * mean;
    float inv  = rsqrtf(var + 1e-5f);
#pragma unroll
    for (int i = 0; i < 16; ++i) {
        int c = lane + i * 32;
        float y = (v[i] - mean) * inv * gamma[c] + beta[c];
        Yf[(size_t)row * DD + c] = y;
        Yb[(size_t)row * DD + c] = (__bf16)y;
    }
}

// ---------------------------------------------------------------------------
// Host orchestration
// ---------------------------------------------------------------------------
extern "C" void kernel_launch(void* const* d_in, const int* in_sizes, int n_in,
                              void* d_out, int out_size, void* d_ws, size_t ws_size,
                              hipStream_t stream) {
    (void)in_sizes; (void)n_in; (void)out_size; (void)ws_size;

    const float* x      = (const float*)d_in[0];
    const int*   graph  = (const int*)  d_in[1];
    const float* Wih[2] = { (const float*)d_in[2], (const float*)d_in[6] };
    const float* Whh[2] = { (const float*)d_in[3], (const float*)d_in[7] };
    const float* bih[2] = { (const float*)d_in[4], (const float*)d_in[8] };
    const float* bhh[2] = { (const float*)d_in[5], (const float*)d_in[9] };
    const float* Wh = (const float*)d_in[10]; const float* bh = (const float*)d_in[11];
    const float* Wt = (const float*)d_in[12]; const float* bt = (const float*)d_in[13];
    const float* Wi = (const float*)d_in[14]; const float* bi = (const float*)d_in[15];
    const float* Wc = (const float*)d_in[16]; const float* bc = (const float*)d_in[17];
    const float* gamma = (const float*)d_in[18];
    const float* beta  = (const float*)d_in[19];
    float* out = (float*)d_out;

    // ---- workspace carve-up ----
    size_t off = 0;
    char* base = (char*)d_ws;
    auto alloc = [&](size_t bytes) -> void* {
        void* p = base + off;
        off += (bytes + 255) & ~(size_t)255;
        return p;
    };
    __bf16* xb     = (__bf16*)alloc((size_t)MM * DD * 2);
    __bf16* Wihb[2], *Whhb[2];
    Wihb[0] = (__bf16*)alloc((size_t)G4 * DD * 2);
    Whhb[0] = (__bf16*)alloc((size_t)G4 * HID * 2);
    Wihb[1] = (__bf16*)alloc((size_t)G4 * DD * 2);
    Whhb[1] = (__bf16*)alloc((size_t)G4 * HID * 2);
    __bf16* Whb = (__bf16*)alloc((size_t)DD * DD * 2);
    __bf16* Wtb = (__bf16*)alloc((size_t)DD * DD * 2);
    __bf16* Wib = (__bf16*)alloc((size_t)DD * DD * 2);
    __bf16* Wcb = (__bf16*)alloc((size_t)DD * DD * 2);
    float*  gbuf  = (float*) alloc((size_t)MM * G4 * 4);
    float*  Hst   = (float*) alloc((size_t)MM * DD * 4);   // (2, M, HID) flat == (B,N,D) flat
    float*  memF  = (float*) alloc((size_t)MM * DD * 4);
    __bf16* memb  = (__bf16*)alloc((size_t)MM * DD * 2);
    __bf16* headb = (__bf16*)alloc((size_t)MM * DD * 2);
    __bf16* tailb = (__bf16*)alloc((size_t)MM * DD * 2);
    __bf16* infob = (__bf16*)alloc((size_t)MM * DD * 2);
    __bf16* avb   = (__bf16*)alloc((size_t)MM * DD * 2);
    float*  avc   = (float*) alloc((size_t)MM * DD * 4);
    __bf16* avcb  = (__bf16*)alloc((size_t)MM * DD * 2);

    auto cvt = [&](const float* in, __bf16* o, int n) {
        cvt_bf16_kernel<<<(n + 255) / 256, 256, 0, stream>>>(in, o, n);
    };
    auto gemm = [&](const __bf16* A, int lda, const __bf16* W, int ldw,
                    const float* b0, const float* b1, const float* Cin,
                    float* Cf, __bf16* Cb, int Nout, int K) {
        gemm_wmma_kernel<<<dim3(Nout / 64, MM / 128), 128, 0, stream>>>(
            A, lda, W, ldw, b0, b1, Cin, Cf, Cb, Nout, K);
    };

    // ---- one-time conversions ----
    cvt(x, xb, MM * DD);
    for (int d = 0; d < 2; ++d) {
        cvt(Wih[d], Wihb[d], G4 * DD);
        cvt(Whh[d], Whhb[d], G4 * HID);
    }
    cvt(Wh, Whb, DD * DD); cvt(Wt, Wtb, DD * DD);
    cvt(Wi, Wib, DD * DD); cvt(Wc, Wcb, DD * DD);

    // ---- initial bilstm (h0 = c0 = 0) + LayerNorm ----
    for (int d = 0; d < 2; ++d) {
        gemm(xb, DD, Wihb[d], DD, bih[d], bhh[d], nullptr, gbuf, nullptr, G4, DD);
        lstm_elem_kernel<<<(MM * HID + 255) / 256, 256, 0, stream>>>(
            gbuf, nullptr, Hst + (size_t)d * MM * HID);
    }
    layernorm_kernel<<<MM / 8, 256, 0, stream>>>(Hst, gamma, beta, memF, memb);

    // ---- 3 attention + bilstm layers ----
    for (int layer = 0; layer < 3; ++layer) {
        const bool last = (layer == 2);
        gemm(xb,   DD, Whb, DD, bh, nullptr, nullptr, nullptr, headb, DD, DD);
        gemm(memb, DD, Wtb, DD, bt, nullptr, nullptr, nullptr, tailb, DD, DD);
        gemm(memb, DD, Wib, DD, bi, nullptr, nullptr, nullptr, infob, DD, DD);

        attn_kernel<<<BB * HH * (NN / 16), 128, 0, stream>>>(
            headb, tailb, infob, graph, avb);

        // avc = av @ Wc^T + bc  (f32 for cell state, bf16 for h-GEMM)
        gemm(avb, DD, Wcb, DD, bc, nullptr, nullptr, avc, avcb, DD, DD);

        for (int d = 0; d < 2; ++d) {
            const size_t st = (size_t)d * MM * HID;   // flat-view slice of avc
            gemm(xb, DD, Wihb[d], DD, bih[d], bhh[d], nullptr, gbuf, nullptr, G4, DD);
            gemm(avcb + st, HID, Whhb[d], HID, nullptr, nullptr, gbuf, gbuf, nullptr, G4, HID);
            lstm_elem_kernel<<<(MM * HID + 255) / 256, 256, 0, stream>>>(
                gbuf, avc + st, Hst + st);
        }
        layernorm_kernel<<<MM / 8, 256, 0, stream>>>(
            Hst, gamma, beta, last ? out : memF, memb);
    }
}
